// TriangleMultiplicativeUpdate_12455405158977
// MI455X (gfx1250) — matrix operand order repaired and checked by reading, not compiled, not run on hardware
//
#include <hip/hip_runtime.h>
#include <hip/hip_bf16.h>
#include <math.h>

typedef __bf16 bf16_t;
typedef __attribute__((ext_vector_type(16))) __bf16 v16bf;
typedef __attribute__((ext_vector_type(8)))  __bf16 v8bf;
typedef __attribute__((ext_vector_type(8)))  float  v8f;

#define NDIM 384
#define NN   (384 * 384)
#define CDIM 128

union Frag {
    v16bf v;
    v8bf  h[2];
};

__device__ __forceinline__ float sigmoidf_(float x) { return 1.0f / (1.0f + __expf(-x)); }

// ---------------------------------------------------------------------------
// k0: shuffle six 128x128 fp32 weight matrices into bf16 WMMA B-fragment
// layout.  Dest flat index t encodes (mat, kt, nt, lane, t16); source element
// is W[k][n] with k = kt*32 + (t16>>3)*16 + (lane>>4)*8 + (t16&7), n = nt*16 + (lane&15).
// After this, a wave's B fragment for (kt, nt) is 16 contiguous halfs per lane.
// ---------------------------------------------------------------------------
__global__ void __launch_bounds__(256)
k0_shuffle_weights(const float* __restrict__ w0, const float* __restrict__ w1,
                   const float* __restrict__ w2, const float* __restrict__ w3,
                   const float* __restrict__ w4, const float* __restrict__ w5,
                   bf16_t* __restrict__ wbuf)
{
    int t = blockIdx.x * blockDim.x + threadIdx.x;   // 0 .. 6*16384-1
    int mat  = t >> 14;
    int rem  = t & 16383;
    int kt   = rem >> 12;
    int nt   = (rem >> 9) & 7;
    int lane = (rem >> 4) & 31;
    int t16  = rem & 15;
    int k = kt * 32 + ((t16 >> 3) << 4) + ((lane >> 4) << 3) + (t16 & 7);
    int n = nt * 16 + (lane & 15);
    const float* W = (mat == 0) ? w0 : (mat == 1) ? w1 : (mat == 2) ? w2
                    : (mat == 3) ? w3 : (mat == 4) ? w4 : w5;
    wbuf[t] = (bf16_t)W[k * CDIM + n];
}

// ---------------------------------------------------------------------------
// k1: fused LayerNorm + five 128x128 projections + gating.
// Block = 128 threads = 4 waves; each wave owns a 16-position M-tile.
// A fragments (zn, bf16) are built once in LDS and reused for all 5 weight
// matrices x 8 N-tiles x 4 K-steps  ->  160 v_wmma_f32_16x16x32_bf16 / wave.
// Outputs: a_t, b_t channel-major [c][pos] bf16 (feeds per-channel GEMM),
//          g position-major [pos][c] bf16 (feeds final gating).
// ---------------------------------------------------------------------------
__global__ void __launch_bounds__(128)
k1_ln_proj(const float* __restrict__ z, const float* __restrict__ mask,
           const float* __restrict__ ln_w, const float* __restrict__ ln_b,
           const float* __restrict__ b_a_g, const float* __restrict__ b_a_p,
           const float* __restrict__ b_b_g, const float* __restrict__ b_b_p,
           const float* __restrict__ b_gg,
           const bf16_t* __restrict__ wbuf,
           bf16_t* __restrict__ a_t, bf16_t* __restrict__ b_t,
           bf16_t* __restrict__ g_out)
{
    __shared__ bf16_t ztile[4][16][CDIM];

    int lane = threadIdx.x & 31;
    int wv   = threadIdx.x >> 5;
    int p0   = blockIdx.x * 64 + wv * 16;
    int row  = lane & 15;
    int half = lane >> 4;
    int c0   = half * 64;
    int p    = p0 + row;

    // ---- LayerNorm over 128 channels (2 lanes per row, 64 channels each) ----
    float zr[64];
    const float* zp = z + (size_t)p * CDIM + c0;
    float s = 0.f, sq = 0.f;
#pragma unroll
    for (int i = 0; i < 64; ++i) { float v = zp[i]; zr[i] = v; s += v; sq += v * v; }
    s  += __shfl_xor(s, 16, 32);
    sq += __shfl_xor(sq, 16, 32);
    float mu  = s * (1.f / 128.f);
    float var = sq * (1.f / 128.f) - mu * mu;
    float rs  = rsqrtf(var + 1e-5f);
#pragma unroll
    for (int i = 0; i < 64; ++i) {
        float v = (zr[i] - mu) * rs * ln_w[c0 + i] + ln_b[c0 + i];
        ztile[wv][row][c0 + i] = (bf16_t)v;
    }
    __syncthreads();

    // ---- gather A fragments (ISA 16-bit A 16x32 lane layout) ----
    Frag afr[4];
#pragma unroll
    for (int kt = 0; kt < 4; ++kt) {
        int o1 = kt * 32 + half * 8;
        int o2 = kt * 32 + 16 + half * 8;
        afr[kt].h[0] = *(const v8bf*)&ztile[wv][row][o1];
        afr[kt].h[1] = *(const v8bf*)&ztile[wv][row][o2];
    }

#pragma unroll 1
    for (int nt = 0; nt < 8; ++nt) {
        v8f acc_ag = {}; v8f acc_ap = {}; v8f acc_bg = {}; v8f acc_bp = {}; v8f acc_g = {};
#pragma unroll
        for (int kt = 0; kt < 4; ++kt) {
            size_t fo = (size_t)((kt * 8 + nt) * 32 + lane) * 16;
            v16bf w_ag = *(const v16bf*)(wbuf + 0 * 16384 + fo);
            v16bf w_ap = *(const v16bf*)(wbuf + 1 * 16384 + fo);
            v16bf w_bg = *(const v16bf*)(wbuf + 2 * 16384 + fo);
            v16bf w_bp = *(const v16bf*)(wbuf + 3 * 16384 + fo);
            v16bf w_g  = *(const v16bf*)(wbuf + 4 * 16384 + fo);
            acc_ag = __builtin_amdgcn_wmma_f32_16x16x32_bf16(false, afr[kt].v, false, w_ag, (short)0, acc_ag, false, false);
            acc_ap = __builtin_amdgcn_wmma_f32_16x16x32_bf16(false, afr[kt].v, false, w_ap, (short)0, acc_ap, false, false);
            acc_bg = __builtin_amdgcn_wmma_f32_16x16x32_bf16(false, afr[kt].v, false, w_bg, (short)0, acc_bg, false, false);
            acc_bp = __builtin_amdgcn_wmma_f32_16x16x32_bf16(false, afr[kt].v, false, w_bp, (short)0, acc_bp, false, false);
            acc_g  = __builtin_amdgcn_wmma_f32_16x16x32_bf16(false, afr[kt].v, false, w_g,  (short)0, acc_g,  false, false);
        }
        // ---- epilogue: bias + sigmoid gating + scattered bf16 stores ----
        int n = nt * 16 + (lane & 15);
        float bag = b_a_g[n], bap = b_a_p[n], bbg = b_b_g[n], bbp = b_b_p[n], bgg = b_gg[n];
#pragma unroll
        for (int r = 0; r < 8; ++r) {
            int m  = half * 8 + r;          // D-tile: lanes>=16 hold rows 8..15
            int pp = p0 + m;
            float mv = mask[pp];
            float av = mv * sigmoidf_(acc_ag[r] + bag) * (acc_ap[r] + bap);
            float bv = mv * sigmoidf_(acc_bg[r] + bbg) * (acc_bp[r] + bbp);
            float gv = sigmoidf_(acc_g[r] + bgg);
            a_t[(size_t)n * NN + pp]   = (bf16_t)av;   // channel-major
            b_t[(size_t)n * NN + pp]   = (bf16_t)bv;   // channel-major
            g_out[(size_t)pp * CDIM + n] = (bf16_t)gv; // position-major
        }
    }
}

// ---------------------------------------------------------------------------
// k2: triangle einsum  x[i,j,c] = sum_k a[i,k,c]*b[j,k,c]
// = per-channel GEMM  X_c = A_c * B_c^T  on channel-major bf16 tensors.
// One wave per (c, i-tile, j-strip of 64): the A fragment is loaded once per
// K-step and reused across 4 WMMAs (4 j-subtiles) -> 40 B loaded per lane per
// WMMA instead of 64 B, and 4x fewer A loads.  K = 384 -> 12 K-steps.
// ---------------------------------------------------------------------------
__global__ void __launch_bounds__(256)
k2_triangle(const bf16_t* __restrict__ a_t, const bf16_t* __restrict__ b_t,
            float* __restrict__ x)
{
    int lane = threadIdx.x & 31;
    int wv   = threadIdx.x >> 5;
    int wg   = blockIdx.x * 8 + wv;          // 0 .. 128*24*6-1
    int c    = wg / 144;                     // channel
    int rem  = wg - c * 144;
    int it   = rem / 6;                      // i-tile (16 rows)
    int j4   = rem - it * 6;                 // j-strip (64 cols = 4 subtiles)
    int half = lane >> 4;
    int l15  = lane & 15;

    const bf16_t* chA = a_t + (size_t)c * NN;
    const bf16_t* chB = b_t + (size_t)c * NN;
    const bf16_t* A  = chA + (size_t)(it * 16 + l15) * NDIM;
    const bf16_t* B0 = chB + (size_t)(j4 * 64 +  0 + l15) * NDIM;
    const bf16_t* B1 = chB + (size_t)(j4 * 64 + 16 + l15) * NDIM;
    const bf16_t* B2 = chB + (size_t)(j4 * 64 + 32 + l15) * NDIM;
    const bf16_t* B3 = chB + (size_t)(j4 * 64 + 48 + l15) * NDIM;

    v8f acc0 = {}, acc1 = {}, acc2 = {}, acc3 = {};
#pragma unroll 1
    for (int kt = 0; kt < 12; ++kt) {
        Frag af, bf0, bf1, bf2, bf3;
        int oa = kt * 32 + half * 16;
        af.h[0]  = *(const v8bf*)(A + oa);
        af.h[1]  = *(const v8bf*)(A + oa + 8);
        int ob = kt * 32 + half * 16;          // B^T: lane n=j reads its own row of b
        bf0.h[0] = *(const v8bf*)(B0 + ob);
        bf0.h[1] = *(const v8bf*)(B0 + ob + 8);
        bf1.h[0] = *(const v8bf*)(B1 + ob);
        bf1.h[1] = *(const v8bf*)(B1 + ob + 8);
        bf2.h[0] = *(const v8bf*)(B2 + ob);
        bf2.h[1] = *(const v8bf*)(B2 + ob + 8);
        bf3.h[0] = *(const v8bf*)(B3 + ob);
        bf3.h[1] = *(const v8bf*)(B3 + ob + 8);
        if (kt < 11) {                          // global_prefetch next K-step
            __builtin_prefetch(A + oa + 32, 0, 0);
            __builtin_prefetch(B0 + ob + 32, 0, 0);
            __builtin_prefetch(B2 + ob + 32, 0, 0);
        }
        acc0 = __builtin_amdgcn_wmma_f32_16x16x32_bf16(false, af.v, false, bf0.v, (short)0, acc0, false, false);
        acc1 = __builtin_amdgcn_wmma_f32_16x16x32_bf16(false, af.v, false, bf1.v, (short)0, acc1, false, false);
        acc2 = __builtin_amdgcn_wmma_f32_16x16x32_bf16(false, af.v, false, bf2.v, (short)0, acc2, false, false);
        acc3 = __builtin_amdgcn_wmma_f32_16x16x32_bf16(false, af.v, false, bf3.v, (short)0, acc3, false, false);
    }
#pragma unroll
    for (int r = 0; r < 8; ++r) {
        int m = half * 8 + r;
        size_t rowbase = ((size_t)(it * 16 + m) * NDIM + j4 * 64 + l15) * CDIM + c;
        x[rowbase +  0 * (size_t)CDIM] = acc0[r];
        x[rowbase + 16 * (size_t)CDIM] = acc1[r];
        x[rowbase + 32 * (size_t)CDIM] = acc2[r];
        x[rowbase + 48 * (size_t)CDIM] = acc3[r];
    }
}

// ---------------------------------------------------------------------------
// k3: LayerNorm(x) over C, project by w_z (WMMA), add b_z, gate by g.
// Same wave/tile structure as k1.
// ---------------------------------------------------------------------------
__global__ void __launch_bounds__(128)
k3_lnout_proj(const float* __restrict__ x,
              const float* __restrict__ ln_w, const float* __restrict__ ln_b,
              const float* __restrict__ b_z,
              const bf16_t* __restrict__ wz_frag,   // wbuf + 5*16384
              const bf16_t* __restrict__ g_in,
              float* __restrict__ out)
{
    __shared__ bf16_t xt[4][16][CDIM];

    int lane = threadIdx.x & 31;
    int wv   = threadIdx.x >> 5;
    int p0   = blockIdx.x * 64 + wv * 16;
    int row  = lane & 15;
    int half = lane >> 4;
    int c0   = half * 64;
    int p    = p0 + row;

    float xr[64];
    const float* xp = x + (size_t)p * CDIM + c0;
    float s = 0.f, sq = 0.f;
#pragma unroll
    for (int i = 0; i < 64; ++i) { float v = xp[i]; xr[i] = v; s += v; sq += v * v; }
    s  += __shfl_xor(s, 16, 32);
    sq += __shfl_xor(sq, 16, 32);
    float mu  = s * (1.f / 128.f);
    float var = sq * (1.f / 128.f) - mu * mu;
    float rs  = rsqrtf(var + 1e-5f);
#pragma unroll
    for (int i = 0; i < 64; ++i) {
        float v = (xr[i] - mu) * rs * ln_w[c0 + i] + ln_b[c0 + i];
        xt[wv][row][c0 + i] = (bf16_t)v;
    }
    __syncthreads();

    Frag afr[4];
#pragma unroll
    for (int kt = 0; kt < 4; ++kt) {
        int o1 = kt * 32 + half * 8;
        int o2 = kt * 32 + 16 + half * 8;
        afr[kt].h[0] = *(const v8bf*)&xt[wv][row][o1];
        afr[kt].h[1] = *(const v8bf*)&xt[wv][row][o2];
    }

#pragma unroll 1
    for (int nt = 0; nt < 8; ++nt) {
        v8f acc = {};
#pragma unroll
        for (int kt = 0; kt < 4; ++kt) {
            size_t fo = (size_t)((kt * 8 + nt) * 32 + lane) * 16;
            v16bf w = *(const v16bf*)(wz_frag + fo);
            acc = __builtin_amdgcn_wmma_f32_16x16x32_bf16(false, afr[kt].v, false, w, (short)0, acc, false, false);
        }
        int n  = nt * 16 + (lane & 15);
        float bz = b_z[n];
#pragma unroll
        for (int r = 0; r < 8; ++r) {
            int m  = half * 8 + r;
            int pp = p0 + m;
            float gv = (float)g_in[(size_t)pp * CDIM + n];
            out[(size_t)pp * CDIM + n] = (acc[r] + bz) * gv;
        }
    }
}

// ---------------------------------------------------------------------------
// Host side
// ---------------------------------------------------------------------------
extern "C" void kernel_launch(void* const* d_in, const int* in_sizes, int n_in,
                              void* d_out, int out_size, void* d_ws, size_t ws_size,
                              hipStream_t stream)
{
    const float* z      = (const float*)d_in[0];
    const float* mask   = (const float*)d_in[1];
    const float* ln_in_w  = (const float*)d_in[2];
    const float* ln_in_b  = (const float*)d_in[3];
    const float* w_a_g  = (const float*)d_in[4];
    const float* b_a_g  = (const float*)d_in[5];
    const float* w_a_p  = (const float*)d_in[6];
    const float* b_a_p  = (const float*)d_in[7];
    const float* w_b_g  = (const float*)d_in[8];
    const float* b_b_g  = (const float*)d_in[9];
    const float* w_b_p  = (const float*)d_in[10];
    const float* b_b_p  = (const float*)d_in[11];
    const float* ln_out_w = (const float*)d_in[12];
    const float* ln_out_b = (const float*)d_in[13];
    const float* w_z    = (const float*)d_in[14];
    const float* b_z    = (const float*)d_in[15];
    const float* w_g    = (const float*)d_in[16];
    const float* b_g    = (const float*)d_in[17];
    float* out = (float*)d_out;

    // workspace layout (bytes): ~189 MB total
    char* ws = (char*)d_ws;
    const size_t OFF_WB = 0;                                   // 6*16384 bf16
    const size_t OFF_A  = OFF_WB + 6 * 16384 * sizeof(bf16_t); // 128*NN bf16
    const size_t OFF_B  = OFF_A + (size_t)CDIM * NN * sizeof(bf16_t);
    const size_t OFF_G  = OFF_B + (size_t)CDIM * NN * sizeof(bf16_t);
    const size_t OFF_X  = OFF_G + (size_t)NN * CDIM * sizeof(bf16_t);

    bf16_t* wbuf = (bf16_t*)(ws + OFF_WB);
    bf16_t* a_t  = (bf16_t*)(ws + OFF_A);
    bf16_t* b_t  = (bf16_t*)(ws + OFF_B);
    bf16_t* gbuf = (bf16_t*)(ws + OFF_G);
    float*  xbuf = (float*)(ws + OFF_X);

    // k0: weight shuffle (mat order: a_g, a_p, b_g, b_p, g, z)
    k0_shuffle_weights<<<384, 256, 0, stream>>>(w_a_g, w_a_p, w_b_g, w_b_p, w_g, w_z, wbuf);

    // k1: fused LN + 5 projections  (NN/64 blocks of 4 waves)
    k1_ln_proj<<<NN / 64, 128, 0, stream>>>(z, mask, ln_in_w, ln_in_b,
                                            b_a_g, b_a_p, b_b_g, b_b_p, b_g,
                                            wbuf, a_t, b_t, gbuf);

    // k2: 128 per-channel GEMMs, one wave per (c, it, j-strip) -> 18432 waves / 8
    k2_triangle<<<(CDIM * 24 * 6) / 8, 256, 0, stream>>>(a_t, b_t, xbuf);

    // k3: LN over C + w_z projection + gating
    k3_lnout_proj<<<NN / 64, 128, 0, stream>>>(xbuf, ln_out_w, ln_out_b, b_z,
                                               wbuf + 5 * 16384, gbuf, out);

    (void)in_sizes; (void)n_in; (void)out_size; (void)ws_size;
}
